// VolterraLayer2D_Lossless_12945031430253
// MI455X (gfx1250) — matrix operand order repaired and checked by reading, not compiled, not run on hardware
//
#include <hip/hip_runtime.h>

typedef __attribute__((ext_vector_type(16))) _Float16 v16h;
typedef __attribute__((ext_vector_type(8)))  float    v8f;

#define BB   8
#define CIN  64
#define COUT 64
#define HH   256
#define WW   256

#define AFRAG_HALVES (144 * 512)          // 4 m-tiles * 36 k-chunks * 32 lanes * 16 halves
#define AFRAG_BYTES  (AFRAG_HALVES * 2)   // 147456 B of packed A, LDS-resident per block

// ---------------------------------------------------------------------------
// Kernel 1: quad = sum_shifts clip(x * roll(x), -1, 1) (circular rolls),
// repack x and quad to f16 NHWC (channel contiguous) via LDS transpose.
// ---------------------------------------------------------------------------
__global__ __launch_bounds__(256) void pack_acts(const float* __restrict__ x,
                                                 _Float16* __restrict__ xh,
                                                 _Float16* __restrict__ qh) {
  __shared__ _Float16 lx[64 * 66];   // [w_local][ci], padded against bank conflicts
  __shared__ _Float16 lq[64 * 66];
  int blk = blockIdx.x;              // 8*256*4 blocks
  int wt = blk & 3;
  int h  = (blk >> 2) & 255;
  int b  = blk >> 10;
  int w0 = wt * 64;
  int t  = threadIdx.x;
  int hm = (h - 1) & 255, hp = (h + 1) & 255;
  const float* xb = x + (size_t)b * CIN * HH * WW;

  #pragma unroll 4
  for (int r = 0; r < 16; ++r) {
    int idx = t + 256 * r;           // 0..4095
    int ci  = idx >> 6;              // reads coalesced along w
    int wl  = idx & 63;
    int w   = w0 + wl;
    int wm  = (w - 1) & 255;
    const float* xc = xb + (size_t)ci * HH * WW;
    float c   = xc[h  * WW + w];
    float up  = xc[hm * WW + w];     // shift (s1=0,s2=1)  -> x[h-1, w]
    float dlr = xc[hp * WW + wm];    // shift (1,-1)       -> x[h+1, w-1]
    float lf  = xc[h  * WW + wm];    // shift (1,0)        -> x[h,   w-1]
    float ul  = xc[hm * WW + wm];    // shift (1,1)        -> x[h-1, w-1]
    float q = fminf(fmaxf(c * up,  -1.f), 1.f)
            + fminf(fmaxf(c * dlr, -1.f), 1.f)
            + fminf(fmaxf(c * lf,  -1.f), 1.f)
            + fminf(fmaxf(c * ul,  -1.f), 1.f);
    lx[wl * 66 + ci] = (_Float16)c;
    lq[wl * 66 + ci] = (_Float16)q;
  }
  __syncthreads();

  size_t obase = ((size_t)(b * HH + h) * WW + w0) * (size_t)CIN;
  #pragma unroll 4
  for (int r = 0; r < 16; ++r) {
    int idx = t + 256 * r;
    int wl  = idx >> 6;
    int ci  = idx & 63;              // writes coalesced along ci (128B lines)
    xh[obase + (size_t)wl * CIN + ci] = lx[wl * 66 + ci];
    qh[obase + (size_t)wl * CIN + ci] = lq[wl * 66 + ci];
  }
}

// ---------------------------------------------------------------------------
// Kernel 2: pack [w1 | w2] (f32, OIHW) into f16 A-fragments in a
// contiguous-k 16x32 layout (k = 16*hi + e), not the ISA table order; B matches.
// K = src*576 + tap*64 + ci, 36 K-chunks of 32, 4 M-tiles of 16.
// apack[((mt*36+kc)*32 + lane)*16 + e]
// ---------------------------------------------------------------------------
__global__ void pack_w(const float* __restrict__ w1,
                       const float* __restrict__ w2,
                       _Float16* __restrict__ apack) {
  int id = blockIdx.x * 256 + threadIdx.x;
  if (id >= AFRAG_HALVES) return;
  int e    = id & 15;
  int lane = (id >> 4) & 31;
  int frag = id >> 9;                // 0..143
  int kc   = frag % 36;
  int mt   = frag / 36;
  int hi   = lane >> 4;
  int m    = mt * 16 + (lane & 15);
  int j = e >> 1, hhh = e & 1;
  int kloc = hi * 16 + j * 2 + hhh;
  int K   = kc * 32 + kloc;
  int src = K / 576;
  int rem = K % 576;
  int tap = rem >> 6;
  int ci  = rem & 63;
  const float* w = src ? w2 : w1;
  apack[id] = (_Float16)w[(m * 64 + ci) * 9 + tap];
}

// ---------------------------------------------------------------------------
// Kernel 3: fused dual-conv GEMM. Wave tile: 64 (C_out) x 32 (spatial).
// A (147 KB) staged into LDS per block with gfx1250 async direct-to-LDS
// loads (no VGPR round-trip), shared by 8 waves; B gathered from L2-resident
// f16 NHWC; 36 chunks x 4 m-tiles x 2 n-tiles = 288 WMMA/wave.
// ---------------------------------------------------------------------------
__global__ __launch_bounds__(256) void conv_wmma(const _Float16* __restrict__ xh,
                                                 const _Float16* __restrict__ qh,
                                                 const _Float16* __restrict__ apack,
                                                 const float* __restrict__ b1,
                                                 const float* __restrict__ b2,
                                                 float* __restrict__ out) {
  extern __shared__ _Float16 a_lds[];          // AFRAG_HALVES

  // Async stage of packed A into LDS: GLOBAL_LOAD_ASYNC_TO_LDS_B128
  // (memory -> LDS, tracked by ASYNCcnt; low 32 bits of a generic LDS
  // pointer are the LDS byte offset per the aperture scheme).
  {
    unsigned ldsbase = (unsigned)(uintptr_t)a_lds;
    const char* g = (const char*)apack;
    #pragma unroll 4
    for (int i = threadIdx.x; i < AFRAG_BYTES / 16; i += 256) {
      unsigned loff = ldsbase + (unsigned)i * 16u;
      unsigned long long ga = (unsigned long long)(uintptr_t)(g + (size_t)i * 16);
      asm volatile("global_load_async_to_lds_b128 %0, %1, off"
                   :: "v"(loff), "v"(ga) : "memory");
    }
    asm volatile("s_wait_asynccnt 0x0" ::: "memory");
  }
  __syncthreads();

  int wave = (blockIdx.x << 3) + (threadIdx.x >> 5);  // 16384 waves
  int lane = threadIdx.x & 31;
  int wt = wave & 7;                // 32-wide W strip
  int h  = (wave >> 3) & 255;
  int b  = wave >> 11;
  int n  = lane & 15;
  int hi = lane >> 4;
  int w0 = wt * 32 + n;             // n-tile 0 column; n-tile 1 is w0+16
  int ci0 = hi * 16;                // 16 contiguous ci per lane-half

  v8f acc[8];
  #pragma unroll
  for (int i = 0; i < 8; ++i) acc[i] = (v8f){};

  #pragma unroll
  for (int s = 0; s < 2; ++s) {
    const _Float16* sp = s ? qh : xh;
    #pragma unroll
    for (int tap = 0; tap < 9; ++tap) {
      int dh = tap / 3 - 1, dw = tap % 3 - 1;
      int y   = h + dh;
      int xx0 = w0 + dw;            // n-tile 0 source column (per lane)
      int xx1 = xx0 + 16;           // n-tile 1 source column
      bool yin  = (unsigned)y   < (unsigned)HH;
      bool inb0 = yin && ((unsigned)xx0 < (unsigned)WW);
      bool inb1 = yin && ((unsigned)xx1 < (unsigned)WW);
      size_t rbase = ((size_t)(b * HH + y) * WW) * (size_t)CIN;
      #pragma unroll
      for (int half = 0; half < 2; ++half) {
        size_t coff = (size_t)half * 32 + ci0;
        v16h bf0 = {}, bf1 = {};                     // zero pad at borders
        if (inb0) bf0 = *(const v16h*)(sp + rbase + (size_t)xx0 * CIN + coff);
        if (inb1) bf1 = *(const v16h*)(sp + rbase + (size_t)xx1 * CIN + coff);
        int kc = s * 18 + tap * 2 + half;
        const _Float16* ap = a_lds + ((size_t)kc * 32 + lane) * 16;
        #pragma unroll
        for (int mt = 0; mt < 4; ++mt) {
          v16h a = *(const v16h*)(ap + (size_t)mt * 36 * 512);
          acc[mt * 2 + 0] = __builtin_amdgcn_wmma_f32_16x16x32_f16(
              false, a, false, bf0, (short)0, acc[mt * 2 + 0], false, false);
          acc[mt * 2 + 1] = __builtin_amdgcn_wmma_f32_16x16x32_f16(
              false, a, false, bf1, (short)0, acc[mt * 2 + 1], false, false);
        }
      }
    }
  }

  // Epilogue: C/D layout -> m = r + 8*hi within m-tile, n = lane%16.
  #pragma unroll
  for (int mt = 0; mt < 4; ++mt) {
    #pragma unroll
    for (int t2 = 0; t2 < 2; ++t2) {
      v8f cc = acc[mt * 2 + t2];
      #pragma unroll
      for (int r = 0; r < 8; ++r) {
        int co = mt * 16 + hi * 8 + r;
        float bias = b1[co] + 4.0f * b2[co];
        out[(((size_t)b * COUT + co) * HH + h) * WW + w0 + t2 * 16] = cc[r] + bias;
      }
    }
  }
}

// ---------------------------------------------------------------------------
extern "C" void kernel_launch(void* const* d_in, const int* in_sizes, int n_in,
                              void* d_out, int out_size, void* d_ws, size_t ws_size,
                              hipStream_t stream) {
  const float* x  = (const float*)d_in[0];
  const float* w1 = (const float*)d_in[1];
  const float* b1 = (const float*)d_in[2];
  const float* w2 = (const float*)d_in[3];
  const float* b2 = (const float*)d_in[4];
  float* out = (float*)d_out;

  const size_t actBytes = (size_t)BB * HH * WW * CIN * sizeof(_Float16); // 64 MiB
  char* ws = (char*)d_ws;
  _Float16* xh = (_Float16*)ws;
  _Float16* qh = (_Float16*)(ws + actBytes);
  _Float16* ap = (_Float16*)(ws + 2 * actBytes);

  pack_acts<<<BB * HH * 4, 256, 0, stream>>>(x, xh, qh);
  pack_w<<<(AFRAG_HALVES + 255) / 256, 256, 0, stream>>>(w1, w2, ap);

  int conv_blocks = (BB * HH * (WW / 32)) / 8;   // 16384 waves / 8 per block
  conv_wmma<<<conv_blocks, 256, AFRAG_BYTES, stream>>>(xh, qh, ap, b1, b2, out);
}
